// Matching_Attention_807453852063
// MI455X (gfx1250) — matrix operand order repaired and checked by reading, not compiled, not run on hardware
//
#include <hip/hip_runtime.h>
#include <hip/hip_bf16.h>

typedef __bf16 bf16_t;
typedef __attribute__((ext_vector_type(16))) __bf16 v16bf;
typedef __attribute__((ext_vector_type(8)))  __bf16 v8bf;
typedef __attribute__((ext_vector_type(4)))  __bf16 v4bf;
typedef __attribute__((ext_vector_type(8)))  float  v8f;

#define TILE_M 128
#define TILE_N 128
#define TILE_K 32
#define LDSK   40   // padded LDS K-stride (elements) to avoid bank conflicts

__device__ __forceinline__ v16bf cat8(v8bf lo, v8bf hi) {
  return __builtin_shufflevector(lo, hi, 0,1,2,3,4,5,6,7,8,9,10,11,12,13,14,15);
}

// ---------------------------------------------------------------------------
// Generic batched bf16 GEMM:  C[b] = A[b] * op(B[b]) (+bias)(+relu)
//   A: row-major [M][K]        (lda = K stride in elements)
//   BT=false: B row-major [K][N]   (NN)
//   BT=true : B row-major [N][K]   (i.e. C = A * B^T, NT)
//   EPI: 0 = none, 1 = +bias, 2 = +bias+relu
//   WBF: also write a bf16 copy of C
// Block: 256 threads = 8 waves; block tile 128x128; wave tile 64x32.
// Double-buffered LDS, ONE barrier per K-stage:
//   compute(buf kt&1) ; store stage kt+1 -> buf (kt+1)&1 ; __syncthreads()
//   RAW: barrier at end of iter kt orders stores before compute(kt+1).
//   WAR: barrier at end of iter kt-1 orders compute(kt-1) before iter kt's store.
// Requires M%128==0, N%128==0, K%32==0.
// ---------------------------------------------------------------------------
template<bool BT, int EPI, bool WBF>
__global__ __launch_bounds__(256, 1)
void gemm_bf16_wmma(const bf16_t* __restrict__ A, long long sA, int lda,
                    const bf16_t* __restrict__ B, long long sB, int ldb,
                    const float*  __restrict__ bias,
                    float* __restrict__ C, long long sC, int ldc,
                    bf16_t* __restrict__ Cbf, long long sCbf,
                    int K)
{
  __shared__ bf16_t As[2][TILE_M][LDSK];
  __shared__ bf16_t Bs[2][TILE_N][LDSK];

  const int t    = threadIdx.x;
  const int lane = t & 31;
  const int wid  = t >> 5;
  const int bz   = blockIdx.z;
  const int m0   = blockIdx.x * TILE_M;
  const int n0   = blockIdx.y * TILE_N;

  const bf16_t* __restrict__ Ab = A + (long long)bz * sA;
  const bf16_t* __restrict__ Bb = B + (long long)bz * sB;

  // staging thread mapping
  const int srow = t >> 1;         // 0..127 row (A) / N-row (B when BT)
  const int scol = (t & 1) * 16;   // 0 or 16 within the 32-wide K stage
  const int bk   = t & 31;         // NN: k within stage
  const int bn0  = (t >> 5) * 16;  // NN: 16-wide N segment

  v8bf ra0, ra1, rb0, rb1;

  auto loadA = [&](int k0) {
    const bf16_t* p = Ab + (long long)(m0 + srow) * lda + (k0 + scol);
    ra0 = *(const v8bf*)(p);
    ra1 = *(const v8bf*)(p + 8);
  };
  auto loadB = [&](int k0) {
    if (BT) {
      const bf16_t* p = Bb + (long long)(n0 + srow) * ldb + (k0 + scol);
      rb0 = *(const v8bf*)(p);
      rb1 = *(const v8bf*)(p + 8);
    } else {
      const bf16_t* p = Bb + (long long)(k0 + bk) * ldb + (n0 + bn0);
      rb0 = *(const v8bf*)(p);
      rb1 = *(const v8bf*)(p + 8);
    }
  };
  auto storeStage = [&](int buf) {
    *(v8bf*)&As[buf][srow][scol]     = ra0;
    *(v8bf*)&As[buf][srow][scol + 8] = ra1;
    if (BT) {
      *(v8bf*)&Bs[buf][srow][scol]     = rb0;
      *(v8bf*)&Bs[buf][srow][scol + 8] = rb1;
    } else {
      #pragma unroll
      for (int i = 0; i < 8; ++i) Bs[buf][bn0 + i][bk]     = rb0[i];
      #pragma unroll
      for (int i = 0; i < 8; ++i) Bs[buf][bn0 + 8 + i][bk] = rb1[i];
    }
  };

  // wave tiling: 2 waves over M (64 rows each), 4 waves over N (32 cols each)
  const int wm = (wid & 1) * 64;
  const int wn = (wid >> 1) * 32;
  const int lr = lane & 15;   // row/col within 16x16 tile
  const int kh = lane >> 4;   // lane half selects K-subset

  v8f acc[4][2];
  #pragma unroll
  for (int tm = 0; tm < 4; ++tm)
    #pragma unroll
    for (int tn = 0; tn < 2; ++tn)
      acc[tm][tn] = (v8f)(0.0f);

  const int nStages = K / TILE_K;

  loadA(0); loadB(0);
  storeStage(0);
  __syncthreads();

  for (int kt = 0; kt < nStages; ++kt) {
    const int buf = kt & 1;

    if (kt + 1 < nStages) { loadA((kt + 1) * TILE_K); loadB((kt + 1) * TILE_K); }
    if (kt + 2 < nStages) {  // L2 prefetch two stages ahead (global_prefetch_b8)
      __builtin_prefetch(Ab + (long long)(m0 + srow) * lda + (kt + 2) * TILE_K + scol, 0, 1);
      if (BT)
        __builtin_prefetch(Bb + (long long)(n0 + srow) * ldb + (kt + 2) * TILE_K + scol, 0, 1);
      else
        __builtin_prefetch(Bb + (long long)((kt + 2) * TILE_K + bk) * ldb + (n0 + bn0), 0, 1);
    }

    // A fragments: lane (0-15): M=lr, K = {kh*8..+7} U {16+kh*8..+7}
    v16bf af[4];
    #pragma unroll
    for (int tm = 0; tm < 4; ++tm) {
      const bf16_t* p = &As[buf][wm + tm * 16 + lr][kh * 8];
      v8bf lo = *(const v8bf*)(p);
      v8bf hi = *(const v8bf*)(p + 16);
      af[tm] = cat8(lo, hi);
    }
    // B fragments: lane (0-15): N=lr, K = {kh*8..+7} U {16+kh*8..+7}
    v16bf bfr[2];
    #pragma unroll
    for (int tn = 0; tn < 2; ++tn) {
      const bf16_t* p = &Bs[buf][wn + tn * 16 + lr][kh * 8];
      v8bf lo = *(const v8bf*)(p);
      v8bf hi = *(const v8bf*)(p + 16);
      bfr[tn] = cat8(lo, hi);
    }

    #pragma unroll
    for (int tm = 0; tm < 4; ++tm)
      #pragma unroll
      for (int tn = 0; tn < 2; ++tn)
        acc[tm][tn] = __builtin_amdgcn_wmma_f32_16x16x32_bf16(
            false, af[tm], false, bfr[tn], (short)0, acc[tm][tn], false, false);

    if (kt + 1 < nStages) {
      storeStage((kt + 1) & 1);   // safe: buf (kt+1)&1 last read in compute(kt-1),
      __syncthreads();            // ordered by previous iteration's barrier
    }
  }

  // epilogue: C/D layout — VGPR i: lanes 0-15 -> M=i, lanes 16-31 -> M=8+i; N=lr
  float*  Cb   = C + (long long)bz * sC;
  bf16_t* Cbfb = WBF ? (Cbf + (long long)bz * sCbf) : (bf16_t*)nullptr;
  #pragma unroll
  for (int tm = 0; tm < 4; ++tm) {
    #pragma unroll
    for (int tn = 0; tn < 2; ++tn) {
      const int col = n0 + wn + tn * 16 + lr;
      const float bv = (EPI > 0) ? bias[col] : 0.0f;
      #pragma unroll
      for (int i = 0; i < 8; ++i) {
        const int row = m0 + wm + tm * 16 + kh * 8 + i;
        float v = acc[tm][tn][i] + bv;
        if (EPI == 2) v = v > 0.0f ? v : 0.0f;
        Cb[(long long)row * ldc + col] = v;
        if (WBF) Cbfb[(long long)row * ldc + col] = (bf16_t)v;
      }
    }
  }
}

// ---------------------------------------------------------------------------
// f32 -> bf16 conversion (n must be a multiple of 4)
// ---------------------------------------------------------------------------
__global__ void cvt_f32_bf16(const float* __restrict__ in, bf16_t* __restrict__ out,
                             long long n) {
  long long i = ((long long)blockIdx.x * blockDim.x + threadIdx.x) * 4;
  if (i < n) {
    float4 v = *(const float4*)(in + i);
    v4bf o; o[0] = (bf16_t)v.x; o[1] = (bf16_t)v.y; o[2] = (bf16_t)v.z; o[3] = (bf16_t)v.w;
    *(v4bf*)(out + i) = o;
  }
}

// ---------------------------------------------------------------------------
// Row softmax over A_len=1024, write bf16 W[p][a] and transposed Wt[a][p]
// grid: (P_len, bs), 256 threads, 4 elems/thread
// ---------------------------------------------------------------------------
__global__ __launch_bounds__(256)
void softmax_wt(const float* __restrict__ W, bf16_t* __restrict__ Wbf,
                bf16_t* __restrict__ Wt, int Alen, int Plen) {
  const int p = blockIdx.x, b = blockIdx.y, t = threadIdx.x;
  const float* row = W + ((long long)b * Plen + p) * Alen;
  float4 v = ((const float4*)row)[t];

  __shared__ float red[256];
  red[t] = fmaxf(fmaxf(v.x, v.y), fmaxf(v.z, v.w));
  __syncthreads();
  #pragma unroll
  for (int s = 128; s > 0; s >>= 1) {
    if (t < s) red[t] = fmaxf(red[t], red[t + s]);
    __syncthreads();
  }
  const float rmax = red[0];
  __syncthreads();

  float e0 = __expf(v.x - rmax), e1 = __expf(v.y - rmax);
  float e2 = __expf(v.z - rmax), e3 = __expf(v.w - rmax);
  red[t] = e0 + e1 + e2 + e3;
  __syncthreads();
  #pragma unroll
  for (int s = 128; s > 0; s >>= 1) {
    if (t < s) red[t] += red[t + s];
    __syncthreads();
  }
  const float inv = 1.0f / red[0];

  const int a0 = t * 4;
  v4bf o;
  o[0] = (bf16_t)(e0 * inv); o[1] = (bf16_t)(e1 * inv);
  o[2] = (bf16_t)(e2 * inv); o[3] = (bf16_t)(e3 * inv);
  *(v4bf*)(Wbf + ((long long)b * Plen + p) * Alen + a0) = o;

  bf16_t* wt = Wt + (long long)b * Alen * Plen + p;
  #pragma unroll
  for (int j = 0; j < 4; ++j)
    wt[(long long)(a0 + j) * Plen] = o[j];
}

// ---------------------------------------------------------------------------
// Build bf16 concat input: Out[r][c] = Mx-Base, Out[r][L+c] = Mx*Base
// ---------------------------------------------------------------------------
__global__ void build_pair(const float* __restrict__ Mx, const float* __restrict__ Base,
                           bf16_t* __restrict__ Out, long long total, int L) {
  long long i = (long long)blockIdx.x * blockDim.x + threadIdx.x;
  if (i >= total) return;
  long long r = i / L;
  int c = (int)(i - r * L);
  float m = Mx[i], bse = Base[i];
  bf16_t* o = Out + r * (2LL * L);
  o[c]     = (bf16_t)(m - bse);
  o[c + L] = (bf16_t)(m * bse);
}

// ---------------------------------------------------------------------------
extern "C" void kernel_launch(void* const* d_in, const int* in_sizes, int n_in,
                              void* d_out, int out_size, void* d_ws, size_t ws_size,
                              hipStream_t stream) {
  (void)in_sizes; (void)n_in; (void)out_size; (void)ws_size;
  const int BS = 16, AL = 1024, PL = 2048, L = 768, L2 = 1536;
  const long long nA  = (long long)BS * AL * L;   // 12,582,912
  const long long nP  = (long long)BS * PL * L;   // 25,165,824
  const long long nGw = (long long)L * L;
  const long long nFw = (long long)L2 * L;
  const long long nW  = (long long)BS * PL * AL;  // 33,554,432

  const float* A_f   = (const float*)d_in[0];
  const float* P_f   = (const float*)d_in[1];
  const float* Gw_f  = (const float*)d_in[2];
  const float* Gb    = (const float*)d_in[3];
  const float* fcaw_f= (const float*)d_in[4];
  const float* fcab  = (const float*)d_in[5];
  const float* fcpw_f= (const float*)d_in[6];
  const float* fcpb  = (const float*)d_in[7];

  char* wsp = (char*)d_ws;
  size_t off = 0;
  auto alloc = [&](size_t bytes) -> void* {
    off = (off + 255) & ~(size_t)255;
    void* p = wsp + off;
    off += bytes;
    return p;
  };

  bf16_t* A_bf   = (bf16_t*)alloc(2 * nA);
  bf16_t* P_bf   = (bf16_t*)alloc(2 * nP);
  bf16_t* Gw_bf  = (bf16_t*)alloc(2 * nGw);
  bf16_t* fcaw_bf= (bf16_t*)alloc(2 * nFw);
  bf16_t* fcpw_bf= (bf16_t*)alloc(2 * nFw);
  float*  Ag_f   = (float*) alloc(4 * nA);
  bf16_t* Ag_bf  = (bf16_t*)alloc(2 * nA);
  float*  W_f    = (float*) alloc(4 * nW);   // reused for MP_f32 after softmax
  bf16_t* W_bf   = (bf16_t*)alloc(2 * nW);   // W_bf..Wt contiguous: reused for SPin
  bf16_t* Wt_bf  = (bf16_t*)alloc(2 * nW);
  float*  MA_f   = (float*) alloc(4 * nA);
  bf16_t* SAin   = (bf16_t*)alloc(2 * (long long)BS * AL * L2);
  float*  MP_f   = W_f;                      // alias (W scores consumed by softmax)
  bf16_t* SPin   = W_bf;                     // alias (W_bf/Wt consumed by MP/MA gemms)

  float* outSA = (float*)d_out;
  float* outSP = (float*)d_out + nA;

  const int TB = 256;
  // ---- f32 -> bf16 conversions
  cvt_f32_bf16<<<dim3((unsigned)((nA  / 4 + TB - 1) / TB)), TB, 0, stream>>>(A_f,   A_bf,   nA);
  cvt_f32_bf16<<<dim3((unsigned)((nP  / 4 + TB - 1) / TB)), TB, 0, stream>>>(P_f,   P_bf,   nP);
  cvt_f32_bf16<<<dim3((unsigned)((nGw / 4 + TB - 1) / TB)), TB, 0, stream>>>(Gw_f,  Gw_bf,  nGw);
  cvt_f32_bf16<<<dim3((unsigned)((nFw / 4 + TB - 1) / TB)), TB, 0, stream>>>(fcaw_f,fcaw_bf,nFw);
  cvt_f32_bf16<<<dim3((unsigned)((nFw / 4 + TB - 1) / TB)), TB, 0, stream>>>(fcpw_f,fcpw_bf,nFw);

  // ---- Ag = A @ G_w + G_b   (NN, bias, f32 + bf16 outputs)
  gemm_bf16_wmma<false, 1, true><<<dim3(AL / TILE_M, L / TILE_N, BS), TB, 0, stream>>>(
      A_bf, (long long)AL * L, L, Gw_bf, 0LL, L, Gb,
      Ag_f, (long long)AL * L, L, Ag_bf, (long long)AL * L, L);

  // ---- W = P @ Ag^T   (NT)
  gemm_bf16_wmma<true, 0, false><<<dim3(PL / TILE_M, AL / TILE_N, BS), TB, 0, stream>>>(
      P_bf, (long long)PL * L, L, Ag_bf, (long long)AL * L, L, nullptr,
      W_f, (long long)PL * AL, AL, nullptr, 0LL, L);

  // ---- softmax over A_len; emit bf16 W and W^T
  softmax_wt<<<dim3(PL, BS), TB, 0, stream>>>(W_f, W_bf, Wt_bf, AL, PL);

  // ---- MP = W @ Ag   (NN, K=1024)
  gemm_bf16_wmma<false, 0, false><<<dim3(PL / TILE_M, L / TILE_N, BS), TB, 0, stream>>>(
      W_bf, (long long)PL * AL, AL, Ag_bf, (long long)AL * L, L, nullptr,
      MP_f, (long long)PL * L, L, nullptr, 0LL, AL);

  // ---- MA = W^T @ P  (NN, K=2048)
  gemm_bf16_wmma<false, 0, false><<<dim3(AL / TILE_M, L / TILE_N, BS), TB, 0, stream>>>(
      Wt_bf, (long long)AL * PL, PL, P_bf, (long long)PL * L, L, nullptr,
      MA_f, (long long)AL * L, L, nullptr, 0LL, PL);

  // ---- concat features
  build_pair<<<dim3((unsigned)((nA + TB - 1) / TB)), TB, 0, stream>>>(MA_f, Ag_f, SAin, nA, L);
  build_pair<<<dim3((unsigned)((nP + TB - 1) / TB)), TB, 0, stream>>>(MP_f, P_f,  SPin, nP, L);

  // ---- SA = relu(SAin @ fca_w + fca_b) -> d_out
  gemm_bf16_wmma<false, 2, false><<<dim3(AL / TILE_M, L / TILE_N, BS), TB, 0, stream>>>(
      SAin, (long long)AL * L2, L2, fcaw_bf, 0LL, L, fcab,
      outSA, (long long)AL * L, L, nullptr, 0LL, L2);

  // ---- SP = relu(SPin @ fcp_w + fcp_b) -> d_out (second half)
  gemm_bf16_wmma<false, 2, false><<<dim3(PL / TILE_M, L / TILE_N, BS), TB, 0, stream>>>(
      SPin, (long long)PL * L2, L2, fcpw_bf, 0LL, L, fcpb,
      outSP, (long long)PL * L, L, nullptr, 0LL, L2);
}
